// Attention_66889820668299
// MI455X (gfx1250) — matrix operand order repaired and checked by reading, not compiled, not run on hardware
//
#include <hip/hip_runtime.h>
#include <hip/hip_bf16.h>
#include <stdint.h>

// ---------------------------------------------------------------------------
// CDNA5 (gfx1250) attention block:  out = proj( softmax(QK^T/sqrt(D)) V )
// All matmuls run on v_wmma_f32_16x16x32_bf16 (fp32 accumulate).
// Flash-attention keeps the 512MB score matrix out of memory entirely.
// Softmax row-sum is computed by a WMMA against an all-ones B matrix;
// row-max uses a DPP16 butterfly (no LDS traffic in the reduction).
// ---------------------------------------------------------------------------

typedef __attribute__((ext_vector_type(16))) __bf16          v16bf;
typedef __attribute__((ext_vector_type(8)))  float           v8f;
typedef __attribute__((ext_vector_type(8)))  unsigned short  u16x8;

union Frag {            // 16 bf16 = one WMMA A/B operand (8 VGPRs)
    v16bf bf;
    u16x8 h[2];
};

static constexpr int Bb = 2, Nt = 2048, Cc = 1024, Hh = 16, Dd = 64;

__device__ __forceinline__ unsigned short f2bf(float f) {
    unsigned u = __float_as_uint(f);
    u += 0x7FFFu + ((u >> 16) & 1u);          // round-to-nearest-even
    return (unsigned short)(u >> 16);
}

// DPP16 lane-permute (row-confined; lanes 0-15 and 16-31 stay independent)
template <int CTRL>
__device__ __forceinline__ float dpp_movf(float v) {
    int y = __builtin_amdgcn_update_dpp(0, __float_as_int(v), CTRL, 0xF, 0xF, true);
    return __int_as_float(y);
}
// max across each 16-lane half: xor1, xor2, xor7 (half mirror), xor15 (mirror)
__device__ __forceinline__ float rowmax16(float v) {
    v = fmaxf(v, dpp_movf<0x0B1>(v));   // quad_perm [1,0,3,2]
    v = fmaxf(v, dpp_movf<0x04E>(v));   // quad_perm [2,3,0,1]
    v = fmaxf(v, dpp_movf<0x141>(v));   // row_half_mirror
    v = fmaxf(v, dpp_movf<0x140>(v));   // row_mirror
    return v;
}

__global__ void __launch_bounds__(256)
cvt_f32_bf16(const float* __restrict__ in, unsigned short* __restrict__ out, int n) {
    int i = blockIdx.x * blockDim.x + threadIdx.x;
    if (i < n) out[i] = f2bf(in[i]);
}

// ---------------------------------------------------------------------------
// GEMM:  Out[M,Ncols] = A[M,K](bf16) @ W[Ncols,K]^T (bf16) + bias
// One wave -> 16 rows x 64 cols (4 accumulator tiles), K-loop step 32.
// MODE 0: store fp32 to outF.  MODE 1: scatter bf16 into q / k / v^T layouts
//          (Q is pre-scaled by 1/sqrt(D)=0.125, exact in bf16).
// ---------------------------------------------------------------------------
template <int MODE>
__global__ void __launch_bounds__(256)
gemm_bf16_wmma(const unsigned short* __restrict__ A,
               const unsigned short* __restrict__ W,
               const float* __restrict__ bias,
               float* __restrict__ outF,
               unsigned short* __restrict__ q_us,
               unsigned short* __restrict__ k_us,
               unsigned short* __restrict__ vT_us,
               int K, int Ncols)
{
    const int lane = threadIdx.x & 31;
    const int wv   = threadIdx.x >> 5;
    const int m0   = blockIdx.y * 16;
    const int n0   = (blockIdx.x * 8 + wv) * 64;
    const int l15  = lane & 15;
    const int hiA  = (lane >> 4) << 3;   // A-frag K sub-offset: 0 / 8
    const int hiB  = (lane >> 4) << 4;   // unused: B-frag uses hiA too (0 / 8)

    const size_t arow = (size_t)(m0 + l15) * K;
    v8f acc[4] = {};

    for (int k0 = 0; k0 < K; k0 += 32) {
        Frag a;
        a.h[0] = *(const u16x8*)(A + arow + k0 + hiA);        // K {0..7}/{8..15}
        a.h[1] = *(const u16x8*)(A + arow + k0 + hiA + 16);   // K {16..23}/{24..31}
        #pragma unroll
        for (int t = 0; t < 4; ++t) {
            const size_t wrow = (size_t)(n0 + t * 16 + l15) * K;
            Frag b;
            b.h[0] = *(const u16x8*)(W + wrow + k0 + hiA);        // K {0..7}/{8..15}
            b.h[1] = *(const u16x8*)(W + wrow + k0 + hiA + 16);   // K {16..23}/{24..31}
            acc[t] = __builtin_amdgcn_wmma_f32_16x16x32_bf16(
                false, a.bf, false, b.bf, (short)0, acc[t], false, false);
        }
    }

    const int hiM = (lane >> 4) << 3;     // C-layout: lanes16-31 hold M = r+8
    #pragma unroll
    for (int t = 0; t < 4; ++t) {
        #pragma unroll
        for (int r = 0; r < 8; ++r) {
            const int m   = m0 + r + hiM;
            const int col = n0 + t * 16 + l15;
            const float val = acc[t][r] + bias[col];
            if (MODE == 0) {
                outF[(size_t)m * Ncols + col] = val;
            } else {
                // col -> (q|k|v, head, dim); m -> (batch, token)
                const int sec = col >> 10, cc = col & 1023;
                const int h = cc >> 6, d = cc & 63;
                const int bbi = m >> 11, ntk = m & 2047;
                const size_t bh = (size_t)bbi * Hh + h;
                if (sec == 0)      q_us[(bh * Nt + ntk) * Dd + d] = f2bf(val * 0.125f);
                else if (sec == 1) k_us[(bh * Nt + ntk) * Dd + d] = f2bf(val);
                else               vT_us[(bh * Dd + d) * Nt + ntk] = f2bf(val);  // V^T
            }
        }
    }
}

// ---------------------------------------------------------------------------
// Flash attention: one wave owns 16 query rows, loops 32 keys at a time.
// Per block: S = Q K^T (4 WMMA), online softmax (DPP16 max butterfly),
// P staged bf16 through per-wave LDS tile, row-sum + O via 5 WMMA.
// ---------------------------------------------------------------------------
__global__ void __launch_bounds__(256)
flash_attn_wmma(const unsigned short* __restrict__ q_us,
                const unsigned short* __restrict__ k_us,
                const unsigned short* __restrict__ vT_us,
                unsigned short* __restrict__ attn_us)
{
    __shared__ unsigned short plds[8][16][32];   // per-wave 16x32 bf16 P tile

    const int lane = threadIdx.x & 31;
    const int wv   = threadIdx.x >> 5;
    const int bh   = blockIdx.x;                 // b*H + h
    const int bbi  = bh >> 4;
    const int h    = bh & 15;
    const int m0   = blockIdx.y * 128 + wv * 16;
    const int l15  = lane & 15;
    const int hiA  = (lane >> 4) << 3;
    const int hiB  = (lane >> 4) << 4;
    const int hiM  = (lane >> 4) << 3;

    // Q fragments (Q already pre-scaled by 1/sqrt(D)), loaded once.
    const size_t qbase = ((size_t)bh * Nt + (m0 + l15)) * Dd;
    Frag qf[2];
    #pragma unroll
    for (int c = 0; c < 2; ++c) {
        qf[c].h[0] = *(const u16x8*)(q_us + qbase + c * 32 + hiA);
        qf[c].h[1] = *(const u16x8*)(q_us + qbase + c * 32 + hiA + 16);
    }

    // all-ones bf16 B fragment: P @ ones gives row-sum replicated in all lanes
    Frag onesf;
    #pragma unroll
    for (int i = 0; i < 8; ++i) { onesf.h[0][i] = 0x3F80; onesf.h[1][i] = 0x3F80; }

    v8f o[4] = {};
    v8f lacc = {};                     // softmax denominator accumulator
    float mrow[8];
    #pragma unroll
    for (int r = 0; r < 8; ++r) mrow[r] = -3.0e38f;

    const size_t kbh = (size_t)bh * Nt * Dd;
    const size_t vbh = (size_t)bh * Dd * Nt;

    for (int j0 = 0; j0 < Nt; j0 += 32) {
        // ---- S = Q @ K^T for 16 rows x 32 keys ----
        v8f s[2] = {};
        #pragma unroll
        for (int u = 0; u < 2; ++u) {
            const size_t kr = kbh + (size_t)(j0 + u * 16 + l15) * Dd;
            #pragma unroll
            for (int c = 0; c < 2; ++c) {
                Frag kf;
                kf.h[0] = *(const u16x8*)(k_us + kr + c * 32 + hiA);
                kf.h[1] = *(const u16x8*)(k_us + kr + c * 32 + hiA + 16);
                s[u] = __builtin_amdgcn_wmma_f32_16x16x32_bf16(
                    false, qf[c].bf, false, kf.bf, (short)0, s[u], false, false);
            }
        }

        // prefetch next key block into near (WGP) cache while softmax runs
        if (j0 + 32 < Nt) {
            __builtin_prefetch(k_us + kbh + (size_t)(j0 + 32 + l15) * Dd, 0, 3);
            __builtin_prefetch(vT_us + vbh + (size_t)l15 * Nt + j0 + 32, 0, 3);
        }

        // ---- online softmax: rows split lanes0-15 (M=r) / lanes16-31 (M=r+8) ----
        #pragma unroll
        for (int r = 0; r < 8; ++r) {
            const float s0 = s[0][r];
            const float s1 = s[1][r];
            const float mx   = rowmax16(fmaxf(s0, s1));      // DPP butterfly
            const float mnew = fmaxf(mrow[r], mx);
            const float corr = __expf(mrow[r] - mnew);
            mrow[r] = mnew;
            const float p0 = __expf(s0 - mnew);
            const float p1 = __expf(s1 - mnew);
            lacc[r] = lacc[r] * corr;                        // rescale denominator
            #pragma unroll
            for (int t = 0; t < 4; ++t) o[t][r] = o[t][r] * corr;
            plds[wv][r + hiM][l15]      = f2bf(p0);          // C-layout -> LDS
            plds[wv][r + hiM][l15 + 16] = f2bf(p1);
        }

        // ---- reload P as an A-fragment (in-wave DS ordering; no barrier) ----
        Frag pf;
        pf.h[0] = *(const u16x8*)&plds[wv][l15][hiA];
        pf.h[1] = *(const u16x8*)&plds[wv][l15][hiA + 16];

        // ---- denominator: lacc += P @ ones  (row sum broadcast to all lanes) ----
        lacc = __builtin_amdgcn_wmma_f32_16x16x32_bf16(
            false, pf.bf, false, onesf.bf, (short)0, lacc, false, false);

        // ---- O += P @ V (V pre-transposed: [bh, d, n]) ----
        #pragma unroll
        for (int t = 0; t < 4; ++t) {
            const size_t vr = vbh + (size_t)(t * 16 + l15) * Nt + j0 + hiA;
            Frag vf;
            vf.h[0] = *(const u16x8*)(vT_us + vr);
            vf.h[1] = *(const u16x8*)(vT_us + vr + 16);
            o[t] = __builtin_amdgcn_wmma_f32_16x16x32_bf16(
                false, pf.bf, false, vf.bf, (short)0, o[t], false, false);
        }
    }

    // ---- normalize (fast rcp; output is bf16 anyway) and emit [B,N,C] ----
    float inv[8];
    #pragma unroll
    for (int r = 0; r < 8; ++r) inv[r] = __builtin_amdgcn_rcpf(lacc[r]);
    #pragma unroll
    for (int t = 0; t < 4; ++t) {
        #pragma unroll
        for (int r = 0; r < 8; ++r) {
            const int m = m0 + r + hiM;
            const int d = t * 16 + l15;
            attn_us[((size_t)bbi * Nt + m) * Cc + h * Dd + d] = f2bf(o[t][r] * inv[r]);
        }
    }
}

// ---------------------------------------------------------------------------
extern "C" void kernel_launch(void* const* d_in, const int* in_sizes, int n_in,
                              void* d_out, int out_size, void* d_ws, size_t ws_size,
                              hipStream_t stream)
{
    (void)in_sizes; (void)n_in; (void)out_size; (void)ws_size;

    const float* x      = (const float*)d_in[0];
    const float* qkv_w  = (const float*)d_in[1];
    const float* qkv_b  = (const float*)d_in[2];
    const float* proj_w = (const float*)d_in[3];
    const float* proj_b = (const float*)d_in[4];
    float* out = (float*)d_out;

    char* ws = (char*)d_ws;
    size_t off = 0;
    auto alloc = [&](size_t bytes) -> void* {
        void* p = ws + off;
        off += (bytes + 255) & ~(size_t)255;
        return p;
    };
    unsigned short* x_us     = (unsigned short*)alloc((size_t)Bb * Nt * Cc * 2);      // 8 MB
    unsigned short* qkvw_us  = (unsigned short*)alloc((size_t)3 * Cc * Cc * 2);       // 6 MB
    unsigned short* projw_us = (unsigned short*)alloc((size_t)Cc * Cc * 2);           // 2 MB
    unsigned short* q_us     = (unsigned short*)alloc((size_t)Bb * Hh * Nt * Dd * 2); // 8 MB
    unsigned short* k_us     = (unsigned short*)alloc((size_t)Bb * Hh * Nt * Dd * 2); // 8 MB
    unsigned short* vT_us    = (unsigned short*)alloc((size_t)Bb * Hh * Dd * Nt * 2); // 8 MB
    unsigned short* attn_us  = (unsigned short*)alloc((size_t)Bb * Nt * Cc * 2);      // 8 MB

    // 1) fp32 -> bf16 conversions
    const int nx = Bb * Nt * Cc;
    cvt_f32_bf16<<<(nx + 255) / 256, 256, 0, stream>>>(x, x_us, nx);
    const int nqw = 3 * Cc * Cc;
    cvt_f32_bf16<<<(nqw + 255) / 256, 256, 0, stream>>>(qkv_w, qkvw_us, nqw);
    const int npw = Cc * Cc;
    cvt_f32_bf16<<<(npw + 255) / 256, 256, 0, stream>>>(proj_w, projw_us, npw);

    // 2) QKV projection (M=4096, K=1024, N=3072) with fused q/k/v^T scatter
    dim3 g1((3 * Cc) / 512, (Bb * Nt) / 16);
    gemm_bf16_wmma<1><<<g1, 256, 0, stream>>>(x_us, qkvw_us, qkv_b,
                                              nullptr, q_us, k_us, vT_us,
                                              Cc, 3 * Cc);

    // 3) flash attention: grid = (B*H, N/128), 8 waves * 16 query rows each
    dim3 g2(Bb * Hh, Nt / 128);
    flash_attn_wmma<<<g2, 256, 0, stream>>>(q_us, k_us, vT_us, attn_us);

    // 4) output projection (M=4096, K=1024, N=1024) -> fp32 d_out
    dim3 g3(Cc / 512, (Bb * Nt) / 16);
    gemm_bf16_wmma<0><<<g3, 256, 0, stream>>>(attn_us, projw_us, proj_b,
                                              out, nullptr, nullptr, nullptr,
                                              Cc, Cc);
}
